// StudentNet_40853728920247
// MI455X (gfx1250) — matrix operand order repaired and checked by reading, not compiled, not run on hardware
//
#include <hip/hip_runtime.h>
#include <math.h>

typedef __attribute__((ext_vector_type(16))) _Float16 v16h;
typedef __attribute__((ext_vector_type(8)))  _Float16 v8h;
typedef __attribute__((ext_vector_type(8)))  float    v8f;

#define HW_FULL (2048*2048)

// ---------------------------------------------------------------------------
// Stage 1: bilinear (align_corners=True) downsample of concat(low,med,high)
//          (9,2048,2048) -> (9,128,128)
// ---------------------------------------------------------------------------
__global__ void hdr_downsample9(const float* __restrict__ low,
                                const float* __restrict__ med,
                                const float* __restrict__ high,
                                float* __restrict__ xds) {
  int idx = blockIdx.x * blockDim.x + threadIdx.x;
  if (idx >= 9 * 128 * 128) return;
  int ox = idx & 127;
  int oy = (idx >> 7) & 127;
  int c  = idx >> 14;
  const float* src = (c < 3) ? (low + c * HW_FULL)
                   : (c < 6) ? (med + (c - 3) * HW_FULL)
                             : (high + (c - 6) * HW_FULL);
  const float step = 2047.0f / 127.0f;
  float ys = oy * step, xs = ox * step;
  int y0 = (int)floorf(ys), x0 = (int)floorf(xs);
  float fy = ys - (float)y0, fx = xs - (float)x0;
  int y1 = min(y0 + 1, 2047), x1 = min(x0 + 1, 2047);
  float p00 = src[y0 * 2048 + x0], p01 = src[y0 * 2048 + x1];
  float p10 = src[y1 * 2048 + x0], p11 = src[y1 * 2048 + x1];
  float top = p00 + (p01 - p00) * fx;
  float bot = p10 + (p11 - p10) * fx;
  xds[idx] = top + (bot - top) * fy;
}

// ---------------------------------------------------------------------------
// Stage 2-4: generic 3x3 stride-2 pad-1 conv + ReLU (tiny activations)
// ---------------------------------------------------------------------------
__global__ void hdr_conv3x3s2(const float* __restrict__ in,
                              const float* __restrict__ w,
                              const float* __restrict__ b,
                              float* __restrict__ out,
                              int IC, int OC, int IH, int OH) {
  int idx = blockIdx.x * blockDim.x + threadIdx.x;
  int total = OC * OH * OH;
  if (idx >= total) return;
  int ox = idx % OH;
  int oy = (idx / OH) % OH;
  int oc = idx / (OH * OH);
  float s = b[oc];
  int iy0 = oy * 2 - 1, ix0 = ox * 2 - 1;
  for (int ic = 0; ic < IC; ++ic) {
    const float* ip = in + ic * IH * IH;
    const float* wp = w + (oc * IC + ic) * 9;
#pragma unroll
    for (int ky = 0; ky < 3; ++ky) {
      int iy = iy0 + ky;
      if ((unsigned)iy >= (unsigned)IH) continue;
#pragma unroll
      for (int kx = 0; kx < 3; ++kx) {
        int ix = ix0 + kx;
        if ((unsigned)ix >= (unsigned)IH) continue;
        s += ip[iy * IH + ix] * wp[ky * 3 + kx];
      }
    }
  }
  out[idx] = fmaxf(s, 0.0f);
}

// ---------------------------------------------------------------------------
// Stage 5a: pack act3 (64,256 f32) -> A (256x64 f16, row-major M,K)
//           and w4 (192x64 f32)    -> B (192x64 f16, row-major N,K)
// ---------------------------------------------------------------------------
__global__ void hdr_pack_f16(const float* __restrict__ act3,
                             const float* __restrict__ w4,
                             _Float16* __restrict__ Ah,
                             _Float16* __restrict__ Bh) {
  int idx = blockIdx.x * blockDim.x + threadIdx.x;
  if (idx < 256 * 64) {
    int k = idx & 63, m = idx >> 6;
    Ah[idx] = (_Float16)act3[k * 256 + m];
  } else if (idx < 256 * 64 + 192 * 64) {
    int j = idx - 256 * 64;
    Bh[j] = (_Float16)w4[j];
  }
}

// ---------------------------------------------------------------------------
// Stage 5b: WMMA GEMM  coeff(192x256) = B(192x64) * A(256x64)^T + b4
//   One wave per 16x16 tile. K=64 -> two v_wmma_f32_16x16x32_f16 steps.
//   A fragment (16-bit A 16x32): lane<16 -> K 0..15; lane>=16 -> K 16..31
//   B fragment (16-bit B 32x16, same k order as A): lane<16 -> K 0..15; lane>=16 -> K 16..31
//   C/D: VGPR r holds M=r (lanes 0-15) / M=r+8 (lanes 16-31), N=lane&15
// ---------------------------------------------------------------------------
__global__ __launch_bounds__(32) void hdr_gemm_coeff(
    const _Float16* __restrict__ Ah,   // 256 x 64 (M,K)
    const _Float16* __restrict__ Bh,   // 192 x 64 (N,K)
    const float* __restrict__ b4,      // 192
    float* __restrict__ coeff) {       // 192 x 256 (N,M)
  int mt = blockIdx.x & 15;   // 16 M tiles
  int nt = blockIdx.x >> 4;   // 12 N tiles
  int lane = threadIdx.x;
  int hs = lane >> 4;         // half-wave select
  int lm = lane & 15;
  int m = mt * 16 + lm;
  int n = nt * 16 + lm;
  v8f c = {};
#pragma unroll
  for (int k0 = 0; k0 < 64; k0 += 32) {
    const _Float16* ap = Ah + m * 64 + k0 + hs * 16;
    v8h alo = *(const v8h*)(ap);
    v8h ahi = *(const v8h*)(ap + 8);
    v16h a, b;
#pragma unroll
    for (int i = 0; i < 8; ++i) { a[i] = alo[i]; a[8 + i] = ahi[i]; }
    b = *(const v16h*)(Bh + n * 64 + k0 + hs * 16);
    c = __builtin_amdgcn_wmma_f32_16x16x32_f16(
        /*neg_a=*/false, a, /*neg_b=*/false, b,
        /*c_mod=*/(short)0, c, /*reuse_a=*/false, /*reuse_b=*/false);
  }
#pragma unroll
  for (int r = 0; r < 8; ++r) {
    int M = r + hs * 8;
    int N = nt * 16 + lm;
    coeff[N * 256 + (mt * 16 + M)] = c[r] + b4[N];
  }
}

// ---------------------------------------------------------------------------
// Stage 6: guide, evaluated ONLY at the 64x64 resize sample points
//          (each needs 4 full-res pixels) -- skips the full-res guide pass.
// ---------------------------------------------------------------------------
__device__ __forceinline__ float hdr_guide_at(
    const float* __restrict__ low, const float* __restrict__ med,
    const float* __restrict__ high,
    const float* __restrict__ gw1, const float* __restrict__ gb1,
    const float* __restrict__ gg,  const float* __restrict__ gbeta,
    const float* __restrict__ gmean, const float* __restrict__ gvar,
    const float* __restrict__ gw2, float gb2v, int y, int x) {
  int off = y * 2048 + x;
  float i0 = 0.33f * (low[off] + med[off] + high[off]);
  float i1 = 0.33f * (low[HW_FULL + off] + med[HW_FULL + off] + high[HW_FULL + off]);
  float i2 = 0.33f * (low[2 * HW_FULL + off] + med[2 * HW_FULL + off] + high[2 * HW_FULL + off]);
  float acc = gb2v;
#pragma unroll
  for (int oc = 0; oc < 16; ++oc) {
    float t = gw1[oc * 3 + 0] * i0 + gw1[oc * 3 + 1] * i1 + gw1[oc * 3 + 2] * i2 + gb1[oc];
    t = gg[oc] * (t - gmean[oc]) * rsqrtf(gvar[oc] + 1e-5f) + gbeta[oc];
    t = fmaxf(t, 0.0f);
    acc += gw2[oc] * t;
  }
  return 1.0f / (1.0f + __expf(-acc));
}

__global__ void hdr_guide64(const float* __restrict__ low,
                            const float* __restrict__ med,
                            const float* __restrict__ high,
                            const float* __restrict__ gw1, const float* __restrict__ gb1,
                            const float* __restrict__ gg,  const float* __restrict__ gbeta,
                            const float* __restrict__ gmean, const float* __restrict__ gvar,
                            const float* __restrict__ gw2, const float* __restrict__ gb2,
                            float* __restrict__ guide64) {
  int idx = blockIdx.x * blockDim.x + threadIdx.x;
  if (idx >= 64 * 64) return;
  int j = idx & 63, i = idx >> 6;
  const float step = 2047.0f / 63.0f;
  float ys = i * step, xs = j * step;
  int y0 = (int)floorf(ys), x0 = (int)floorf(xs);
  float fy = ys - (float)y0, fx = xs - (float)x0;
  int y1 = min(y0 + 1, 2047), x1 = min(x0 + 1, 2047);
  float gb2v = gb2[0];
  float g00 = hdr_guide_at(low, med, high, gw1, gb1, gg, gbeta, gmean, gvar, gw2, gb2v, y0, x0);
  float g01 = hdr_guide_at(low, med, high, gw1, gb1, gg, gbeta, gmean, gvar, gw2, gb2v, y0, x1);
  float g10 = hdr_guide_at(low, med, high, gw1, gb1, gg, gbeta, gmean, gvar, gw2, gb2v, y1, x0);
  float g11 = hdr_guide_at(low, med, high, gw1, gb1, gg, gbeta, gmean, gvar, gw2, gb2v, y1, x1);
  float top = g00 + (g01 - g00) * fx;
  float bot = g10 + (g11 - g10) * fx;
  guide64[idx] = top + (bot - top) * fy;
}

// ---------------------------------------------------------------------------
// Stage 7: trilinear slice of the bilateral grid (align_corners=False, zeros)
//   grid channels: x(=W dim)<-guide, y(=H dim)<-hg(i), z(=D dim)<-wg(j)
// ---------------------------------------------------------------------------
__global__ void hdr_slice(const float* __restrict__ coeff,   // (192,16,16) = (12,16D,16H,16W)
                          const float* __restrict__ guide64, // (64,64)
                          float* __restrict__ sc) {          // (12,64,64)
  int idx = blockIdx.x * blockDim.x + threadIdx.x;
  if (idx >= 64 * 64) return;
  int j = idx & 63, i = idx >> 6;
  float gx = guide64[idx];
  float gy = (float)i * (2.0f / 63.0f) - 1.0f;
  float gz = (float)j * (2.0f / 63.0f) - 1.0f;
  float ix = ((gx + 1.0f) * 16.0f - 1.0f) * 0.5f;
  float iy = ((gy + 1.0f) * 16.0f - 1.0f) * 0.5f;
  float iz = ((gz + 1.0f) * 16.0f - 1.0f) * 0.5f;
  int x0 = (int)floorf(ix), y0 = (int)floorf(iy), z0 = (int)floorf(iz);
  float fx = ix - (float)x0, fy = iy - (float)y0, fz = iz - (float)z0;
  float w8[8];
  int   o8[8];
  int q = 0;
#pragma unroll
  for (int dz = 0; dz < 2; ++dz)
#pragma unroll
    for (int dy = 0; dy < 2; ++dy)
#pragma unroll
      for (int dx = 0; dx < 2; ++dx) {
        int zi = z0 + dz, yi = y0 + dy, xi = x0 + dx;
        bool v = (zi >= 0) && (zi < 16) && (yi >= 0) && (yi < 16) && (xi >= 0) && (xi < 16);
        float w = (dz ? fz : 1.0f - fz) * (dy ? fy : 1.0f - fy) * (dx ? fx : 1.0f - fx);
        w8[q] = v ? w : 0.0f;
        int zc = min(max(zi, 0), 15), yc = min(max(yi, 0), 15), xc = min(max(xi, 0), 15);
        o8[q] = zc * 256 + yc * 16 + xc;
        ++q;
      }
  for (int cc = 0; cc < 12; ++cc) {
    const float* vol = coeff + cc * 4096;   // 16*256
    float acc = 0.0f;
#pragma unroll
    for (int t = 0; t < 8; ++t) acc += vol[o8[t]] * w8[t];
    sc[cc * 4096 + idx] = acc;
  }
}

// ---------------------------------------------------------------------------
// Stage 8 (dominant, memory-bound): upsample sc 64->2048 (align_corners=True)
// and apply per-pixel 3x4 affine. One workgroup per row: y-lerped coeff row
// staged in LDS; float4 (b128) global loads/stores for the streaming data.
// ---------------------------------------------------------------------------
__global__ __launch_bounds__(256) void hdr_apply(const float* __restrict__ low,
                                                 const float* __restrict__ med,
                                                 const float* __restrict__ high,
                                                 const float* __restrict__ sc,
                                                 float* __restrict__ out) {
  __shared__ float srow[12 * 64];
  int y = blockIdx.x;
  int tid = threadIdx.x;
  const float step = 63.0f / 2047.0f;
  float ys = y * step;
  int y0 = (int)floorf(ys);
  float fy = ys - (float)y0;
  int y1 = min(y0 + 1, 63);
  for (int t = tid; t < 12 * 64; t += 256) {
    int k = t >> 6, x = t & 63;
    float a = sc[k * 4096 + y0 * 64 + x];
    float b = sc[k * 4096 + y1 * 64 + x];
    srow[t] = a + (b - a) * fy;
  }
  __syncthreads();

  int rowbase = y * 2048;
#pragma unroll
  for (int g = 0; g < 2; ++g) {
    int x4 = tid * 8 + g * 4;
    // CDNA5 speculative prefetch of next row's stream (global_prefetch_b8)
    if (y < 2047) __builtin_prefetch(low + rowbase + 2048 + x4, 0, 1);
    float4 L0 = *(const float4*)(low  + 0 * HW_FULL + rowbase + x4);
    float4 L1 = *(const float4*)(low  + 1 * HW_FULL + rowbase + x4);
    float4 L2 = *(const float4*)(low  + 2 * HW_FULL + rowbase + x4);
    float4 M0 = *(const float4*)(med  + 0 * HW_FULL + rowbase + x4);
    float4 M1 = *(const float4*)(med  + 1 * HW_FULL + rowbase + x4);
    float4 M2 = *(const float4*)(med  + 2 * HW_FULL + rowbase + x4);
    float4 H0 = *(const float4*)(high + 0 * HW_FULL + rowbase + x4);
    float4 H1 = *(const float4*)(high + 1 * HW_FULL + rowbase + x4);
    float4 H2 = *(const float4*)(high + 2 * HW_FULL + rowbase + x4);
    float img0[4], img1[4], img2[4];
    {
      const float *a0 = (const float*)&L0, *b0 = (const float*)&M0, *c0 = (const float*)&H0;
      const float *a1 = (const float*)&L1, *b1 = (const float*)&M1, *c1 = (const float*)&H1;
      const float *a2 = (const float*)&L2, *b2 = (const float*)&M2, *c2 = (const float*)&H2;
#pragma unroll
      for (int p = 0; p < 4; ++p) {
        img0[p] = 0.33f * (a0[p] + b0[p] + c0[p]);
        img1[p] = 0.33f * (a1[p] + b1[p] + c1[p]);
        img2[p] = 0.33f * (a2[p] + b2[p] + c2[p]);
      }
    }
    float o0[4], o1[4], o2[4];
#pragma unroll
    for (int p = 0; p < 4; ++p) {
      int x = x4 + p;
      float xs = x * step;
      int x0 = (int)floorf(xs);
      float fx = xs - (float)x0;
      int x1 = min(x0 + 1, 63);
      float cf[12];
#pragma unroll
      for (int k = 0; k < 12; ++k) {
        float a = srow[k * 64 + x0];
        float b = srow[k * 64 + x1];
        cf[k] = a + (b - a) * fx;
      }
      float p0 = img0[p], p1 = img1[p], p2 = img2[p];
      o0[p] = cf[0] * p0 + cf[1] * p1 + cf[2]  * p2 + cf[3];
      o1[p] = cf[4] * p0 + cf[5] * p1 + cf[6]  * p2 + cf[7];
      o2[p] = cf[8] * p0 + cf[9] * p1 + cf[10] * p2 + cf[11];
    }
    *(float4*)(out + 0 * HW_FULL + rowbase + x4) = make_float4(o0[0], o0[1], o0[2], o0[3]);
    *(float4*)(out + 1 * HW_FULL + rowbase + x4) = make_float4(o1[0], o1[1], o1[2], o1[3]);
    *(float4*)(out + 2 * HW_FULL + rowbase + x4) = make_float4(o2[0], o2[1], o2[2], o2[3]);
  }
}

// ---------------------------------------------------------------------------
// Launch
// ---------------------------------------------------------------------------
extern "C" void kernel_launch(void* const* d_in, const int* in_sizes, int n_in,
                              void* d_out, int out_size, void* d_ws, size_t ws_size,
                              hipStream_t stream) {
  const float* low   = (const float*)d_in[0];
  const float* med   = (const float*)d_in[1];
  const float* high  = (const float*)d_in[2];
  const float* w1    = (const float*)d_in[3];
  const float* b1    = (const float*)d_in[4];
  const float* w2    = (const float*)d_in[5];
  const float* b2    = (const float*)d_in[6];
  const float* w3    = (const float*)d_in[7];
  const float* b3    = (const float*)d_in[8];
  const float* w4    = (const float*)d_in[9];
  const float* b4    = (const float*)d_in[10];
  const float* gw1   = (const float*)d_in[11];
  const float* gb1   = (const float*)d_in[12];
  const float* gg    = (const float*)d_in[13];
  const float* gbeta = (const float*)d_in[14];
  const float* gmean = (const float*)d_in[15];
  const float* gvar  = (const float*)d_in[16];
  const float* gw2   = (const float*)d_in[17];
  const float* gb2   = (const float*)d_in[18];

  float* ws = (float*)d_ws;
  float* xds   = ws + 0;        // 9*128*128      = 147456
  float* act1  = ws + 147456;   // 16*64*64       = 65536
  float* act2  = ws + 212992;   // 32*32*32       = 32768
  float* act3  = ws + 245760;   // 64*16*16       = 16384
  float* coeff = ws + 262144;   // 192*256        = 49152
  float* guide = ws + 311296;   // 64*64          = 4096
  float* sc    = ws + 315392;   // 12*64*64       = 49152
  _Float16* Ah = (_Float16*)(ws + 364544);  // 256*64 halfs (32B aligned)
  _Float16* Bh = (_Float16*)(ws + 372736);  // 192*64 halfs (32B aligned)
  float* out = (float*)d_out;

  hdr_downsample9<<<(9 * 128 * 128 + 255) / 256, 256, 0, stream>>>(low, med, high, xds);
  hdr_conv3x3s2<<<(16 * 64 * 64 + 255) / 256, 256, 0, stream>>>(xds, w1, b1, act1, 9, 16, 128, 64);
  hdr_conv3x3s2<<<(32 * 32 * 32 + 255) / 256, 256, 0, stream>>>(act1, w2, b2, act2, 16, 32, 64, 32);
  hdr_conv3x3s2<<<(64 * 16 * 16 + 255) / 256, 256, 0, stream>>>(act2, w3, b3, act3, 32, 64, 32, 16);
  hdr_pack_f16<<<(256 * 64 + 192 * 64 + 255) / 256, 256, 0, stream>>>(act3, w4, Ah, Bh);
  hdr_gemm_coeff<<<192, 32, 0, stream>>>(Ah, Bh, b4, coeff);
  hdr_guide64<<<16, 256, 0, stream>>>(low, med, high, gw1, gb1, gg, gbeta, gmean, gvar, gw2, gb2, guide);
  hdr_slice<<<16, 256, 0, stream>>>(coeff, guide, sc);
  hdr_apply<<<2048, 256, 0, stream>>>(low, med, high, sc, out);
}
